// MyGraphNetwork_14087492731022
// MI455X (gfx1250) — compile-verified, matrix operand order checked by reading, not run on hardware
//
#include <hip/hip_runtime.h>
#include <hip/hip_bf16.h>
#include <stdint.h>

// Problem constants (match reference)
#define B_   8
#define N_   1024
#define I_   64
#define E_   2
#define H_   2
#define D_   128
#define O_   64
#define EH_  (E_*H_)       // 4
#define EHB_ (EH_*B_)      // 32
#define IN1_ (EH_*D_)      // 512
#define M_   (B_*N_)       // 8192 rows (b,n)

typedef __bf16 bf16x16 __attribute__((ext_vector_type(16)));
typedef float  f32x8   __attribute__((ext_vector_type(8)));

union Frag { bf16x16 v; uint4 q[2]; unsigned short s[16]; };

// gfx1250 async Global->LDS path (ASYNCcnt-tracked), with portable fallback.
#if defined(__has_builtin)
#  if __has_builtin(__builtin_amdgcn_global_load_async_to_lds_b128)
#    define USE_ASYNC_LDS 1
#  endif
#endif
#ifndef USE_ASYNC_LDS
#  define USE_ASYNC_LDS 0
#endif

#if USE_ASYNC_LDS
// Builtin signature (from hipcc diagnostic): first param is
// 'int __attribute__((vector_size(16))) __device__ *' (global AS int4*),
// second is the LDS-side int4*.
typedef int v4i_t __attribute__((vector_size(16)));
typedef __attribute__((address_space(1))) v4i_t* g_v4i_ptr;
typedef __attribute__((address_space(3))) v4i_t* l_v4i_ptr;
#endif

__device__ __forceinline__ void wait_async_all() {
#if USE_ASYNC_LDS
#  if __has_builtin(__builtin_amdgcn_s_wait_asynccnt)
    __builtin_amdgcn_s_wait_asynccnt(0);
#  else
    asm volatile("s_wait_asynccnt 0" ::: "memory");
#  endif
#endif
}

__device__ __forceinline__ unsigned short f2bf(float x) {
    unsigned int u = __float_as_uint(x);
    u += 0x7FFFu + ((u >> 16) & 1u);          // round-to-nearest-even
    return (unsigned short)(u >> 16);
}

// Load a 16x32 bf16 WMMA fragment from a row-major matrix (rows = M for A,
// rows = N-column for B when the matrix is stored transposed/column-major).
// ISA layout: lanes 0-15 hold K 0..7 (VGPR0-3) and 16..23 (VGPR4-7);
// lanes 16-31 hold K 8..15 and 24..31.
__device__ __forceinline__ bf16x16 load_frag(const unsigned short* base, int ld,
                                             int row0, int k0, int lane) {
    int r = lane & 15, half = lane >> 4;
    const unsigned short* p = base + (size_t)(row0 + r) * ld + k0 + half * 8;
    Frag f;
    f.q[0] = *(const uint4*)(p);
    f.q[1] = *(const uint4*)(p + 16);
    return f.v;
}

// Same fragment layout but from an LDS tile stored as [d][32 j's] bf16.
__device__ __forceinline__ bf16x16 lds_frag(const unsigned short* tile, int d0, int lane) {
    int r = lane & 15, half = lane >> 4;
    const unsigned short* p = tile + (d0 + r) * 32 + half * 8;
    Frag f;
    f.q[0] = *(const uint4*)(p);
    f.q[1] = *(const uint4*)(p + 16);
    return f.v;
}

// ---------------------------------------------------------------------------
// Elementwise f32 -> bf16
__global__ void f32_to_bf16_kernel(const float* __restrict__ src,
                                   unsigned short* __restrict__ dst, int n) {
    int tid = blockIdx.x * blockDim.x + threadIdx.x;
    if (tid < n) dst[tid] = f2bf(src[tid]);
}

// Batched transpose + convert: src[nb][R][C] f32 -> dst[nb][C][R] bf16
__global__ void transpose_to_bf16_kernel(const float* __restrict__ src,
                                         unsigned short* __restrict__ dst,
                                         int R, int C, int total) {
    int tid = blockIdx.x * blockDim.x + threadIdx.x;
    if (tid >= total) return;
    int per = R * C;
    int mb = tid / per, rem = tid - mb * per;
    int r = rem / C, c = rem - r * C;
    dst[(size_t)mb * per + (size_t)c * R + r] = f2bf(src[tid]);
}

// ---------------------------------------------------------------------------
// Generic WMMA GEMM: C[M x N] = A[M x K](bf16 row-major) * B (bf16, stored
// transposed: BT[N x K] row-major). Optional per-column bias, ReLU, bf16 or
// f32 output. blockDim = 128 (4 waves); each wave computes one 16x16 tile.
__global__ void wmma_gemm_kernel(const unsigned short* __restrict__ A,
                                 const unsigned short* __restrict__ BT,
                                 const float* __restrict__ bias,
                                 float* __restrict__ Cf,
                                 unsigned short* __restrict__ Cbf,
                                 int M, int N, int K,
                                 long strideBT, long strideC, int act) {
    int lane = threadIdx.x & 31;
    int w    = threadIdx.x >> 5;
    int m0   = blockIdx.x * 16;
    int n0   = blockIdx.y * 64 + w * 16;
    if (n0 >= N) return;
    const unsigned short* bt = BT + (size_t)blockIdx.z * strideBT;

    f32x8 acc = {0.f,0.f,0.f,0.f,0.f,0.f,0.f,0.f};
    for (int k0 = 0; k0 < K; k0 += 32) {
        bf16x16 a = load_frag(A,  K, m0, k0, lane);
        bf16x16 b = load_frag(bt, K, n0, k0, lane);
        acc = __builtin_amdgcn_wmma_f32_16x16x32_bf16(false, a, false, b,
                                                      (short)0, acc, false, false);
    }
    int r15 = lane & 15, half = lane >> 4;
    size_t cbase = (size_t)blockIdx.z * strideC;
#pragma unroll
    for (int rr = 0; rr < 8; ++rr) {
        int row = m0 + rr + half * 8;
        int col = n0 + r15;
        float v = acc[rr];
        if (bias) v += bias[col];
        if (act)  v = v > 0.f ? v : 0.f;
        size_t idx = cbase + (size_t)row * N + col;
        if (Cbf) Cbf[idx] = f2bf(v);
        else     Cf[idx]  = v;
    }
}

// ---------------------------------------------------------------------------
// Per-node attention logits: s1/s2[ehb*N+n] = Wh[ehb*N+n,:] . a{1,2}w[eh] + b
__global__ void att_scores_kernel(const float* __restrict__ Wh,
                                  const float* __restrict__ a1w, const float* __restrict__ a1b,
                                  const float* __restrict__ a2w, const float* __restrict__ a2b,
                                  float* __restrict__ s1, float* __restrict__ s2) {
    int tid = blockIdx.x * blockDim.x + threadIdx.x;   // EHB_*N_ threads
    int eh = tid >> 13;                                 // 8192 rows per (e,h)
    const float* row = Wh + (size_t)tid * D_;
    const float* w1 = a1w + eh * D_;
    const float* w2 = a2w + eh * D_;
    float acc1 = 0.f, acc2 = 0.f;
    for (int d = 0; d < D_; ++d) {
        float v = row[d];
        acc1 += v * w1[d];
        acc2 += v * w2[d];
    }
    s1[tid] = acc1 + a1b[eh];
    s2[tid] = acc2 + a2b[eh];
}

// Column softmax statistics (softmax is over i for each column j).
// Online max/sum so edges are read exactly once in this pass.
__global__ void col_softmax_stats_kernel(const float* __restrict__ edges,
                                         const float* __restrict__ s1,
                                         const float* __restrict__ s2,
                                         float* __restrict__ colmax,
                                         float* __restrict__ colsum) {
    int j   = blockIdx.x * blockDim.x + threadIdx.x;
    int ehb = blockIdx.y;
    int eh = ehb >> 3, b = ehb & 7, e = eh >> 1;
    const float* s1p = s1 + ehb * N_;
    float s2j = s2[ehb * N_ + j];
    const float* ep = edges + (size_t)b * N_ * N_ * E_ + (size_t)j * E_ + e;
    float m = -1e30f, s = 0.f;
    for (int i = 0; i < N_; ++i) {
        float x = s1p[i] + s2j;
        x = x > 0.f ? x : 0.2f * x;                    // leaky_relu(.,0.2)
        x += ep[(size_t)i * (N_ * E_)];
        if (x > m) { s = s * __expf(m - x) + 1.f; m = x; }
        else       { s += __expf(x - m); }
    }
    colmax[ehb * N_ + j] = m;
    colsum[ehb * N_ + j] = s;
}

// WhsT[ehb][d][j] = bf16( Wh[ehb][j][d] / colsum[ehb][j] )  (transposed so
// WMMA B-fragments are contiguous along j).
__global__ void scale_transpose_bf16_kernel(const float* __restrict__ Wh,
                                            const float* __restrict__ colsum,
                                            unsigned short* __restrict__ WhsT) {
    size_t tid = (size_t)blockIdx.x * blockDim.x + threadIdx.x; // EHB*D*N
    int j   = (int)(tid & (N_ - 1));
    int d   = (int)((tid >> 10) & (D_ - 1));
    int ehb = (int)(tid >> 17);
    float v = Wh[((size_t)ehb * N_ + j) * D_ + d] / colsum[ehb * N_ + j];
    WhsT[tid] = f2bf(v);
}

// ---------------------------------------------------------------------------
// Fused pass 2: rebuild P tiles (exp of scores) as bf16 A-fragments and
// contract against scaled Wh^T with WMMA. One wave -> 16 rows x full D=128.
// The 32x128 Wh^T tile for each j-step is staged into LDS (double-buffered,
// async Global->LDS when available) and shared by all 4 waves of the block.
// layer 0: write (msg+sb) as bf16 into concat layout [b,n, eh*D+d]
// layer 1: write elu(msg+sb) as f32 into msg_out[ehb*N+n][d]
__global__ void attn_aggregate_kernel(const float* __restrict__ edges,
                                      const float* __restrict__ s1,
                                      const float* __restrict__ s2,
                                      const float* __restrict__ colmax,
                                      const unsigned short* __restrict__ WhsT,
                                      const float* __restrict__ SB,
                                      unsigned short* __restrict__ concat_out,
                                      float* __restrict__ msg_out,
                                      int layer) {
    __shared__ __align__(16) unsigned short tileB[2][D_ * 32]; // 2 x 8KB
    __shared__ __align__(16) float s2cm[2 * N_];               // {s2, colmax} per j

    int tid  = threadIdx.x;
    int lane = tid & 31;
    int w    = tid >> 5;
    int ehb  = blockIdx.y;
    int eh = ehb >> 3, b = ehb & 7, e = eh >> 1;
    int i0 = (blockIdx.x * 4 + w) * 16;
    int r15 = lane & 15, half = lane >> 4;
    int irow = i0 + r15;

    size_t sj = (size_t)ehb * N_;
    float s1v = s1[sj + irow];
    const float* erowbase = edges + ((size_t)b * N_ + irow) * (N_ * E_);
    const unsigned short* whs = WhsT + (size_t)ehb * D_ * N_;

    // Stage s2/colmax for all 1024 columns into LDS once.
    for (int k = tid; k < N_; k += 128) {
        s2cm[2 * k]     = s2[sj + k];
        s2cm[2 * k + 1] = colmax[sj + k];
    }

    // Cooperative tile stage: 128 d-rows x 32 j's of bf16 = 512 16B chunks.
    auto stage_tile = [&](int buf, int j0) {
#pragma unroll
        for (int it = 0; it < 4; ++it) {
            int c  = tid + it * 128;         // chunk id 0..511
            int d  = c >> 2;                 // 0..127
            int jc = (c & 3) * 8;            // 0,8,16,24
            const unsigned short* g = whs + (size_t)d * N_ + j0 + jc;
            unsigned short* l = &tileB[buf][d * 32 + jc];
#if USE_ASYNC_LDS
            __builtin_amdgcn_global_load_async_to_lds_b128(
                (g_v4i_ptr)(uintptr_t)g,
                (l_v4i_ptr)(unsigned)(uintptr_t)l, 0, 0);
#else
            *(uint4*)l = *(const uint4*)g;
#endif
        }
    };

    stage_tile(0, 0);
    wait_async_all();
    __syncthreads();

    f32x8 acc[8];
#pragma unroll
    for (int t = 0; t < 8; ++t) {
        f32x8 z = {0.f,0.f,0.f,0.f,0.f,0.f,0.f,0.f};
        acc[t] = z;
    }

    for (int jt = 0; jt < N_ / 32; ++jt) {
        int j0  = jt * 32;
        int buf = jt & 1;
        if (jt + 1 < N_ / 32) {
            stage_tile(buf ^ 1, j0 + 32);                 // prefetch next tile
            __builtin_prefetch(erowbase + (size_t)(j0 + 32) * E_, 0, 0);
        }

        // Build P fragment: lanes 0-15 cover j k-chunks {0..7,16..23},
        // lanes 16-31 cover {8..15,24..31}. float4 edge loads give 2 j's
        // (both edge types) per 16B.
        Frag a;
#pragma unroll
        for (int c = 0; c < 2; ++c) {
            int jb = j0 + c * 16 + half * 8;
            const float* eb = erowbase + (size_t)jb * E_;
#pragma unroll
            for (int t = 0; t < 4; ++t) {
                float4 ev = *(const float4*)(eb + t * 4);
                float e0 = e ? ev.y : ev.x;
                float e1 = e ? ev.w : ev.z;
                int j = jb + 2 * t;
                float2 sc0 = *(const float2*)&s2cm[2 * j];
                float2 sc1 = *(const float2*)&s2cm[2 * (j + 1)];
                float x0 = s1v + sc0.x; x0 = x0 > 0.f ? x0 : 0.2f * x0;
                float x1 = s1v + sc1.x; x1 = x1 > 0.f ? x1 : 0.2f * x1;
                a.s[c * 8 + 2 * t]     = f2bf(__expf(x0 + e0 - sc0.y));
                a.s[c * 8 + 2 * t + 1] = f2bf(__expf(x1 + e1 - sc1.y));
            }
        }

#pragma unroll
        for (int dt = 0; dt < 8; ++dt) {
            bf16x16 bf = lds_frag(&tileB[buf][0], dt * 16, lane);
            acc[dt] = __builtin_amdgcn_wmma_f32_16x16x32_bf16(false, a.v, false, bf,
                                                              (short)0, acc[dt],
                                                              false, false);
        }

        wait_async_all();          // own async tile loads landed in LDS
        __syncthreads();           // whole block done writing/reading tiles
    }

    const float* sbp = SB + eh * D_;
#pragma unroll
    for (int dt = 0; dt < 8; ++dt) {
        int d = dt * 16 + r15;
        float sb = sbp[d];
#pragma unroll
        for (int rr = 0; rr < 8; ++rr) {
            int row = i0 + rr + half * 8;
            float v = acc[dt][rr] + sb;
            if (layer == 0) {
                concat_out[((size_t)b * N_ + row) * IN1_ + eh * D_ + d] = f2bf(v);
            } else {
                v = v > 0.f ? v : (__expf(v) - 1.f);   // elu
                msg_out[((size_t)ehb * N_ + row) * D_ + d] = v;
            }
        }
    }
}

// Deterministic mean over the 4 (e,h) heads -> bf16 state for the out GEMM.
__global__ void mean_to_bf16_kernel(const float* __restrict__ msg,
                                    unsigned short* __restrict__ out) {
    int tid = blockIdx.x * blockDim.x + threadIdx.x;  // M_*D_
    int d  = tid & (D_ - 1);
    int bn = tid >> 7;
    float s = 0.f;
#pragma unroll
    for (int eh = 0; eh < EH_; ++eh)
        s += msg[((size_t)eh * M_ + bn) * D_ + d];
    out[tid] = f2bf(0.25f * s);
}

// ---------------------------------------------------------------------------
extern "C" void kernel_launch(void* const* d_in, const int* in_sizes, int n_in,
                              void* d_out, int out_size, void* d_ws, size_t ws_size,
                              hipStream_t stream) {
    const float* nodes = (const float*)d_in[0];
    const float* edges = (const float*)d_in[1];
    const float* Wemb  = (const float*)d_in[2];
    const float* bemb  = (const float*)d_in[3];
    const float* W0    = (const float*)d_in[4];
    const float* A1w0  = (const float*)d_in[5];
    const float* A1b0  = (const float*)d_in[6];
    const float* A2w0  = (const float*)d_in[7];
    const float* A2b0  = (const float*)d_in[8];
    const float* SB0   = (const float*)d_in[9];
    const float* W1    = (const float*)d_in[10];
    const float* A1w1  = (const float*)d_in[11];
    const float* A1b1  = (const float*)d_in[12];
    const float* A2w1  = (const float*)d_in[13];
    const float* A2b1  = (const float*)d_in[14];
    const float* SB1   = (const float*)d_in[15];
    const float* Wout  = (const float*)d_in[16];
    const float* bout  = (const float*)d_in[17];

    // Workspace carve-up (~40 MB total), 256-byte aligned slices.
    char* ws = (char*)d_ws;
    size_t off = 0;
    auto carve = [&](size_t bytes) -> void* {
        void* p = ws + off;
        off = (off + bytes + 255) & ~(size_t)255;
        return p;
    };
    unsigned short* nodes_bf  = (unsigned short*)carve((size_t)M_ * I_ * 2);
    unsigned short* WembT     = (unsigned short*)carve((size_t)D_ * I_ * 2);
    unsigned short* W0T       = (unsigned short*)carve((size_t)EH_ * D_ * D_ * 2);
    unsigned short* W1T       = (unsigned short*)carve((size_t)EH_ * IN1_ * D_ * 2);
    unsigned short* WoutT     = (unsigned short*)carve((size_t)O_ * D_ * 2);
    unsigned short* st_emb    = (unsigned short*)carve((size_t)M_ * D_ * 2);
    unsigned short* concat_bf = (unsigned short*)carve((size_t)M_ * IN1_ * 2);
    float*          Wh        = (float*)carve((size_t)EH_ * M_ * D_ * 4); // reused as msg buf
    unsigned short* WhsT      = (unsigned short*)carve((size_t)EHB_ * D_ * N_ * 2);
    float*          s1        = (float*)carve((size_t)EHB_ * N_ * 4);
    float*          s2        = (float*)carve((size_t)EHB_ * N_ * 4);
    float*          cmax      = (float*)carve((size_t)EHB_ * N_ * 4);
    float*          csum      = (float*)carve((size_t)EHB_ * N_ * 4);
    unsigned short* mean_bf   = (unsigned short*)carve((size_t)M_ * D_ * 2);
    (void)in_sizes; (void)n_in; (void)out_size; (void)ws_size;

    // --- weight / input precision prep ---
    f32_to_bf16_kernel<<<(M_ * I_ + 255) / 256, 256, 0, stream>>>(nodes, nodes_bf, M_ * I_);
    transpose_to_bf16_kernel<<<(I_ * D_ + 255) / 256, 256, 0, stream>>>(Wemb, WembT, I_, D_, I_ * D_);
    transpose_to_bf16_kernel<<<(EH_ * D_ * D_ + 255) / 256, 256, 0, stream>>>(W0, W0T, D_, D_, EH_ * D_ * D_);
    transpose_to_bf16_kernel<<<(EH_ * IN1_ * D_ + 255) / 256, 256, 0, stream>>>(W1, W1T, IN1_, D_, EH_ * IN1_ * D_);
    transpose_to_bf16_kernel<<<(D_ * O_ + 255) / 256, 256, 0, stream>>>(Wout, WoutT, D_, O_, D_ * O_);

    // --- embedding: state = relu(nodes @ Wemb + bemb) -> bf16 ---
    wmma_gemm_kernel<<<dim3(M_ / 16, D_ / 64, 1), 128, 0, stream>>>(
        nodes_bf, WembT, bemb, nullptr, st_emb, M_, D_, I_, 0, 0, 1);

    // --- layer 0 ---
    wmma_gemm_kernel<<<dim3(M_ / 16, D_ / 64, EH_), 128, 0, stream>>>(
        st_emb, W0T, nullptr, Wh, nullptr, M_, D_, D_,
        (long)D_ * D_, (long)M_ * D_, 0);
    att_scores_kernel<<<(EHB_ * N_) / 256, 256, 0, stream>>>(Wh, A1w0, A1b0, A2w0, A2b0, s1, s2);
    col_softmax_stats_kernel<<<dim3(N_ / 256, EHB_), 256, 0, stream>>>(edges, s1, s2, cmax, csum);
    scale_transpose_bf16_kernel<<<(EHB_ * D_ * N_) / 256, 256, 0, stream>>>(Wh, csum, WhsT);
    attn_aggregate_kernel<<<dim3(N_ / 64, EHB_), 128, 0, stream>>>(
        edges, s1, s2, cmax, WhsT, SB0, concat_bf, nullptr, 0);

    // --- layer 1 ---
    wmma_gemm_kernel<<<dim3(M_ / 16, D_ / 64, EH_), 128, 0, stream>>>(
        concat_bf, W1T, nullptr, Wh, nullptr, M_, D_, IN1_,
        (long)IN1_ * D_, (long)M_ * D_, 0);
    att_scores_kernel<<<(EHB_ * N_) / 256, 256, 0, stream>>>(Wh, A1w1, A1b1, A2w1, A2b1, s1, s2);
    col_softmax_stats_kernel<<<dim3(N_ / 256, EHB_), 256, 0, stream>>>(edges, s1, s2, cmax, csum);
    scale_transpose_bf16_kernel<<<(EHB_ * D_ * N_) / 256, 256, 0, stream>>>(Wh, csum, WhsT);
    attn_aggregate_kernel<<<dim3(N_ / 64, EHB_), 128, 0, stream>>>(
        edges, s1, s2, cmax, WhsT, SB1, nullptr, Wh /* msg written in place */, 1);

    // --- head mean + output projection ---
    mean_to_bf16_kernel<<<(M_ * D_) / 256, 256, 0, stream>>>(Wh, mean_bf);
    wmma_gemm_kernel<<<dim3(M_ / 16, 1, 1), 128, 0, stream>>>(
        mean_bf, WoutT, bout, (float*)d_out, nullptr, M_, O_, D_, 0, 0, 0);
}
